// LinearCrossAttentionAdd_86500641342073
// MI455X (gfx1250) — matrix operand order repaired and checked by reading, not compiled, not run on hardware
//
#include <hip/hip_runtime.h>
#include <hip/hip_bf16.h>

// ---------------------------------------------------------------------------
// CDNA5 (gfx1250) wave32 WMMA bf16 pipeline for LinearCrossAttentionAdd.
// D = A x B + C with V_WMMA_F32_16X16X32_BF16, f32 accumulate.
// ---------------------------------------------------------------------------

typedef __bf16 bf16_t;
typedef __attribute__((ext_vector_type(16))) __bf16 v16bf;
typedef __attribute__((ext_vector_type(8)))  __bf16 v8bf;
typedef __attribute__((ext_vector_type(8)))  float  v8f;

#define N_HEADS   8
#define D_HEAD    32
#define BATCH     16
#define CDIM      256
#define NSP       4096            // 64*64 spatial
#define LED       128
#define GN_EPS    1e-5f

// --- WMMA fragment helpers (ISA 7.12.2 layouts, wave32) --------------------
// A (16x32 bf16): lane r=lane&15 holds row M=r; element j<8 -> K=8*hi+j,
// j>=8 -> K=16+8*hi+(j-8).  Two contiguous 16B loads.
__device__ __forceinline__ v16bf load_frag_a(const bf16_t* __restrict__ A,
                                             int lda, int lane) {
  const int r = lane & 15, h = lane >> 4;
  const bf16_t* p = A + (size_t)r * lda + 8 * h;
  v8bf lo = *(const v8bf*)p;
  v8bf hi = *(const v8bf*)(p + 16);
  return __builtin_shufflevector(lo, hi, 0,1,2,3,4,5,6,7,8,9,10,11,12,13,14,15);
}
// B (32x16 bf16): lane holds column N=lane&15; element j<8 -> K=8*hi+j,
// j>=8 -> K=16+8*hi+(j-8).
// Memory is N-major with K contiguous (ldb = K stride between N rows).
__device__ __forceinline__ v16bf load_frag_b(const bf16_t* __restrict__ B,
                                             int ldb, int lane) {
  const int n = lane & 15, h = lane >> 4;
  const bf16_t* p = B + (size_t)n * ldb + 8 * h;
  v8bf lo = *(const v8bf*)p;
  v8bf hi = *(const v8bf*)(p + 16);
  return __builtin_shufflevector(lo, hi, 0,1,2,3,4,5,6,7,8,9,10,11,12,13,14,15);
}
__device__ __forceinline__ v8f wmma_bf16(v16bf a, v16bf b, v8f c) {
  return __builtin_amdgcn_wmma_f32_16x16x32_bf16(false, a, false, b,
                                                 (short)0, c, false, false);
}

// --- K0a: convert w_qkv f32 -> bf16 ---------------------------------------
__global__ void convert_wqkv_kernel(const float* __restrict__ w,
                                    bf16_t* __restrict__ o, int n) {
  int i = blockIdx.x * blockDim.x + threadIdx.x;
  if (i < n) o[i] = (bf16_t)w[i];
}

// --- K0b: lk/lq = condition @ w.T + b  (16x256 each, trivial) -------------
__global__ __launch_bounds__(256)
void lin_kernel(const float* __restrict__ cond,
                const float* __restrict__ wlk, const float* __restrict__ blk,
                const float* __restrict__ wlq, const float* __restrict__ blq,
                float* __restrict__ lk, float* __restrict__ lq) {
  int t = blockIdx.x * 256 + threadIdx.x;     // 0 .. 8191
  int sel = t >> 12;                          // 0 = lk, 1 = lq
  int b = (t >> 8) & 15;
  int c = t & 255;
  const float* W    = sel ? wlq : wlk;
  const float* bias = sel ? blq : blk;
  const float* cb = cond + b * LED;
  const float* wr = W + (size_t)c * LED;
  float s = bias[c];
  for (int j = 0; j < LED; ++j) s += cb[j] * wr[j];
  (sel ? lq : lk)[b * CDIM + c] = s;
}

// --- K1: transpose x[b][c][n] f32 -> xT[b][n][c] bf16 (LDS tiled) ---------
__global__ __launch_bounds__(256)
void transpose_kernel(const float* __restrict__ x, bf16_t* __restrict__ xT) {
  __shared__ float tile[32][33];
  const int b  = blockIdx.z;
  const int c0 = blockIdx.y * 32;             // gridDim.y = 8
  const int n0 = blockIdx.x * 32;             // gridDim.x = 128
  const int tx = threadIdx.x & 31;
  const int ty = threadIdx.x >> 5;            // 0..7
  const float* xb = x + (size_t)b * CDIM * NSP;
  #pragma unroll
  for (int r = 0; r < 32; r += 8)
    tile[ty + r][tx] = xb[(size_t)(c0 + ty + r) * NSP + n0 + tx];
  __syncthreads();
  bf16_t* xTb = xT + (size_t)b * NSP * CDIM;
  #pragma unroll
  for (int r = 0; r < 32; r += 8)
    xTb[(size_t)(n0 + ty + r) * CDIM + c0 + tx] = (bf16_t)tile[tx][ty + r];
}

// --- K2: QKV GEMM. 128 thr = 4 waves, each wave 32x32, block 64x64. -------
// Epilogue: q rows -> qRaw[b][n][c] f32 (+lq, transposed for later B-side),
//           k rows -> kRaw[b][c][n] f32 (+lk), v rows -> vBf[b][c][n] bf16.
__global__ __launch_bounds__(128)
void qkv_gemm_kernel(const bf16_t* __restrict__ Wq, const bf16_t* __restrict__ xT,
                     const float* __restrict__ lq, const float* __restrict__ lk,
                     float* __restrict__ qRaw, float* __restrict__ kRaw,
                     bf16_t* __restrict__ vBf) {
  const int b    = blockIdx.z;
  const int lane = threadIdx.x & 31;
  const int w    = threadIdx.x >> 5;
  const int mbase = blockIdx.y * 64 + (w >> 1) * 32;   // 0..736
  const int nbase = blockIdx.x * 64 + (w & 1) * 32;    // 0..4064
  const bf16_t* xTb = xT + (size_t)b * NSP * CDIM;

  v8f acc[2][2] = {};
  #pragma unroll
  for (int k0 = 0; k0 < CDIM; k0 += 32) {
    __builtin_prefetch(xTb + (size_t)nbase * CDIM + ((k0 + 32) & 255), 0, 0);
    v16bf a0 = load_frag_a(Wq + (size_t)mbase * CDIM + k0, CDIM, lane);
    v16bf a1 = load_frag_a(Wq + (size_t)(mbase + 16) * CDIM + k0, CDIM, lane);
    v16bf b0 = load_frag_b(xTb + (size_t)nbase * CDIM + k0, CDIM, lane);
    v16bf b1 = load_frag_b(xTb + (size_t)(nbase + 16) * CDIM + k0, CDIM, lane);
    acc[0][0] = wmma_bf16(a0, b0, acc[0][0]);
    acc[0][1] = wmma_bf16(a0, b1, acc[0][1]);
    acc[1][0] = wmma_bf16(a1, b0, acc[1][0]);
    acc[1][1] = wmma_bf16(a1, b1, acc[1][1]);
  }

  const int ln = lane & 15, lh = lane >> 4;
  #pragma unroll
  for (int ti = 0; ti < 2; ++ti) {
    const int row0 = mbase + 16 * ti;
    const int s  = row0 >> 8;        // 0=q 1=k 2=v (16-row tile never crosses)
    const int c0 = row0 & 255;
    #pragma unroll
    for (int tj = 0; tj < 2; ++tj) {
      const int n = nbase + 16 * tj + ln;
      if (s == 0) {                  // q, transposed store, +lq
        float* dst = qRaw + ((size_t)b * NSP + n) * CDIM + c0 + 8 * lh;
        const float* lqp = lq + b * CDIM + c0 + 8 * lh;
        #pragma unroll
        for (int v2 = 0; v2 < 8; ++v2) dst[v2] = acc[ti][tj][v2] + lqp[v2];
      } else if (s == 1) {           // k, +lk
        #pragma unroll
        for (int v2 = 0; v2 < 8; ++v2) {
          const int c = c0 + 8 * lh + v2;
          kRaw[((size_t)b * CDIM + c) * NSP + n] = acc[ti][tj][v2] + lk[b * CDIM + c];
        }
      } else {                       // v -> bf16
        #pragma unroll
        for (int v2 = 0; v2 < 8; ++v2) {
          const int c = c0 + 8 * lh + v2;
          vBf[((size_t)b * CDIM + c) * NSP + n] = (bf16_t)acc[ti][tj][v2];
        }
      }
    }
  }
}

// --- K3: q softmax over d=32 (contiguous), * 1/sqrt(32), -> bf16 ----------
__global__ __launch_bounds__(256)
void softmax_q_kernel(const float* __restrict__ qRaw, bf16_t* __restrict__ qSm) {
  int t = blockIdx.x * 256 + threadIdx.x;     // (b*4096+n)*8 + h ; 524288 total
  int h = t & 7;
  int bn = t >> 3;
  const float* p = qRaw + (size_t)bn * CDIM + h * D_HEAD;
  bf16_t*      o = qSm  + (size_t)bn * CDIM + h * D_HEAD;
  float vals[D_HEAD];
  float mx = -3.4e38f;
  #pragma unroll
  for (int d = 0; d < D_HEAD; ++d) { vals[d] = p[d]; mx = fmaxf(mx, vals[d]); }
  float s = 0.f;
  #pragma unroll
  for (int d = 0; d < D_HEAD; ++d) { vals[d] = __expf(vals[d] - mx); s += vals[d]; }
  const float inv = 0.17677669529663689f / s;  // scale = 32^-0.5
  #pragma unroll
  for (int d = 0; d < D_HEAD; ++d) o[d] = (bf16_t)(vals[d] * inv);
}

// --- K4: k softmax over n=4096 (one block per (b,c) row) -> bf16 ----------
__global__ __launch_bounds__(256)
void softmax_k_kernel(const float* __restrict__ kRaw, bf16_t* __restrict__ kSm) {
  __shared__ float red[256];
  const int bc = blockIdx.x;                  // b*256 + c
  const float* p = kRaw + (size_t)bc * NSP;
  bf16_t*      o = kSm  + (size_t)bc * NSP;
  const int t = threadIdx.x;
  float mx = -3.4e38f;
  for (int i = t; i < NSP; i += 256) mx = fmaxf(mx, p[i]);
  red[t] = mx; __syncthreads();
  for (int off = 128; off > 0; off >>= 1) {
    if (t < off) red[t] = fmaxf(red[t], red[t + off]);
    __syncthreads();
  }
  mx = red[0]; __syncthreads();
  float s = 0.f;
  for (int i = t; i < NSP; i += 256) s += __expf(p[i] - mx);
  red[t] = s; __syncthreads();
  for (int off = 128; off > 0; off >>= 1) {
    if (t < off) red[t] += red[t + off];
    __syncthreads();
  }
  const float inv = 1.0f / red[0];
  for (int i = t; i < NSP; i += 256) o[i] = (bf16_t)(__expf(p[i] - mx) * inv);
}

// --- K5: context[b,h] = k_sm @ v^T (32x32, K=4096). 4 waves split K. ------
__global__ __launch_bounds__(128)
void context_kernel(const bf16_t* __restrict__ kSm, const bf16_t* __restrict__ vBf,
                    float* __restrict__ ctx) {
  __shared__ float red[4][32][32];
  const int bh = blockIdx.x;                  // 128 blocks
  const int b = bh >> 3, h = bh & 7;
  const int lane = threadIdx.x & 31;
  const int w    = threadIdx.x >> 5;
  const size_t base = ((size_t)b * CDIM + h * D_HEAD) * NSP;
  v8f acc[2][2] = {};
  for (int n0 = w * 1024; n0 < (w + 1) * 1024; n0 += 32) {
    v16bf a0 = load_frag_a(kSm + base + n0, NSP, lane);
    v16bf a1 = load_frag_a(kSm + base + (size_t)16 * NSP + n0, NSP, lane);
    v16bf b0 = load_frag_b(vBf + base + n0, NSP, lane);
    v16bf b1 = load_frag_b(vBf + base + (size_t)16 * NSP + n0, NSP, lane);
    acc[0][0] = wmma_bf16(a0, b0, acc[0][0]);
    acc[0][1] = wmma_bf16(a0, b1, acc[0][1]);
    acc[1][0] = wmma_bf16(a1, b0, acc[1][0]);
    acc[1][1] = wmma_bf16(a1, b1, acc[1][1]);
  }
  #pragma unroll
  for (int ti = 0; ti < 2; ++ti)
    #pragma unroll
    for (int tj = 0; tj < 2; ++tj)
      #pragma unroll
      for (int v2 = 0; v2 < 8; ++v2)
        red[w][lane][(ti * 2 + tj) * 8 + v2] = acc[ti][tj][v2];
  __syncthreads();
  if (w == 0) {
    const int ln = lane & 15, lh = lane >> 4;
    #pragma unroll
    for (int ti = 0; ti < 2; ++ti)
      #pragma unroll
      for (int tj = 0; tj < 2; ++tj)
        #pragma unroll
        for (int v2 = 0; v2 < 8; ++v2) {
          const int idx = (ti * 2 + tj) * 8 + v2;
          float s = red[0][lane][idx] + red[1][lane][idx]
                  + red[2][lane][idx] + red[3][lane][idx];
          const int d = 16 * ti + 8 * lh + v2;     // row of D
          const int e = 16 * tj + ln;              // col of D
          ctx[(((size_t)b * N_HEADS + h) * D_HEAD + d) * D_HEAD + e] = s;
        }
  }
}

// --- K6: W2[b] = w_out x blockdiag(context_b)  -> bf16 --------------------
// W2[b][o][h*32+d] = sum_e w_out[o][h*32+e] * ctx[b][h][d][e]
__global__ __launch_bounds__(256)
void w2_kernel(const float* __restrict__ wOut, const float* __restrict__ ctx,
               bf16_t* __restrict__ W2) {
  int t = blockIdx.x * 256 + threadIdx.x;     // 16*256*256 = 1M
  int c = t & 255;
  int o = (t >> 8) & 255;
  int b = t >> 16;
  int h = c >> 5, d = c & 31;
  const float* wr = wOut + (size_t)o * CDIM + h * D_HEAD;
  const float* cr = ctx + (((size_t)b * N_HEADS + h) * D_HEAD + d) * D_HEAD;
  float s = 0.f;
  #pragma unroll
  for (int e = 0; e < D_HEAD; ++e) s += wr[e] * cr[e];
  W2[((size_t)b * CDIM + o) * CDIM + c] = (bf16_t)s;
}

// --- K7: y = W2[b] @ q_sm + b_out; per-block (sum, sumsq) partials --------
__global__ __launch_bounds__(128)
void out_gemm_kernel(const bf16_t* __restrict__ W2, const bf16_t* __restrict__ qSm,
                     const float* __restrict__ bOut, float* __restrict__ y,
                     float* __restrict__ partial) {
  __shared__ float s1[128], s2[128];
  const int b    = blockIdx.z;
  const int lane = threadIdx.x & 31;
  const int w    = threadIdx.x >> 5;
  const int mbase = blockIdx.y * 64 + (w >> 1) * 32;   // 0..224
  const int nbase = blockIdx.x * 64 + (w & 1) * 32;
  const bf16_t* A = W2 + (size_t)b * CDIM * CDIM;
  const bf16_t* B = qSm + (size_t)b * NSP * CDIM;
  v8f acc[2][2] = {};
  #pragma unroll
  for (int k0 = 0; k0 < CDIM; k0 += 32) {
    v16bf a0 = load_frag_a(A + (size_t)mbase * CDIM + k0, CDIM, lane);
    v16bf a1 = load_frag_a(A + (size_t)(mbase + 16) * CDIM + k0, CDIM, lane);
    v16bf b0 = load_frag_b(B + (size_t)nbase * CDIM + k0, CDIM, lane);
    v16bf b1 = load_frag_b(B + (size_t)(nbase + 16) * CDIM + k0, CDIM, lane);
    acc[0][0] = wmma_bf16(a0, b0, acc[0][0]);
    acc[0][1] = wmma_bf16(a0, b1, acc[0][1]);
    acc[1][0] = wmma_bf16(a1, b0, acc[1][0]);
    acc[1][1] = wmma_bf16(a1, b1, acc[1][1]);
  }
  const int ln = lane & 15, lh = lane >> 4;
  float ls = 0.f, lsq = 0.f;
  #pragma unroll
  for (int ti = 0; ti < 2; ++ti)
    #pragma unroll
    for (int tj = 0; tj < 2; ++tj)
      #pragma unroll
      for (int v2 = 0; v2 < 8; ++v2) {
        const int c = mbase + 16 * ti + 8 * lh + v2;
        const int n = nbase + 16 * tj + ln;
        const float val = acc[ti][tj][v2] + bOut[c];
        y[((size_t)b * CDIM + c) * NSP + n] = val;
        ls += val; lsq += val * val;
      }
  s1[threadIdx.x] = ls; s2[threadIdx.x] = lsq;
  __syncthreads();
  for (int off = 64; off > 0; off >>= 1) {
    if ((int)threadIdx.x < off) {
      s1[threadIdx.x] += s1[threadIdx.x + off];
      s2[threadIdx.x] += s2[threadIdx.x + off];
    }
    __syncthreads();
  }
  if (threadIdx.x == 0) {
    const int slot = (blockIdx.z * gridDim.y + blockIdx.y) * gridDim.x + blockIdx.x;
    partial[2 * slot]     = s1[0];
    partial[2 * slot + 1] = s2[0];
  }
}

// --- K8: per-batch mean / rstd from 256 partials --------------------------
__global__ __launch_bounds__(256)
void stats_kernel(const float* __restrict__ partial, float* __restrict__ stats) {
  __shared__ float s1[256], s2[256];
  const int b = blockIdx.x, t = threadIdx.x;
  s1[t] = partial[2 * (b * 256 + t)];
  s2[t] = partial[2 * (b * 256 + t) + 1];
  __syncthreads();
  for (int off = 128; off > 0; off >>= 1) {
    if (t < off) { s1[t] += s1[t + off]; s2[t] += s2[t + off]; }
    __syncthreads();
  }
  if (t == 0) {
    const float invN = 1.0f / (float)(CDIM * NSP);
    const float mean = s1[0] * invN;
    const float var  = s2[0] * invN - mean * mean;
    stats[2 * b]     = mean;
    stats[2 * b + 1] = rsqrtf(var + GN_EPS);
  }
}

// --- K9: normalize + affine -> d_out --------------------------------------
__global__ __launch_bounds__(256)
void norm_kernel(const float* __restrict__ y, const float* __restrict__ stats,
                 const float* __restrict__ gamma, const float* __restrict__ beta,
                 float* __restrict__ out) {
  const size_t i = (size_t)blockIdx.x * 256 + threadIdx.x;  // 16,777,216 total
  const int b = (int)(i >> 20);                             // 256*4096 = 2^20
  const int c = (int)(i >> 12) & 255;
  const float mean = stats[2 * b], rstd = stats[2 * b + 1];
  out[i] = (y[i] - mean) * rstd * gamma[c] + beta[c];
}

// ---------------------------------------------------------------------------
extern "C" void kernel_launch(void* const* d_in, const int* in_sizes, int n_in,
                              void* d_out, int out_size, void* d_ws, size_t ws_size,
                              hipStream_t stream) {
  (void)in_sizes; (void)n_in; (void)out_size; (void)ws_size;
  const float* x     = (const float*)d_in[0];
  const float* cond  = (const float*)d_in[1];
  const float* w_qkv = (const float*)d_in[2];
  const float* w_lk  = (const float*)d_in[3];
  const float* b_lk  = (const float*)d_in[4];
  const float* w_lq  = (const float*)d_in[5];
  const float* b_lq  = (const float*)d_in[6];
  const float* w_out = (const float*)d_in[7];
  const float* b_out = (const float*)d_in[8];
  const float* gamma = (const float*)d_in[9];
  const float* beta  = (const float*)d_in[10];
  float* out = (float*)d_out;

  char* ws = (char*)d_ws;
  size_t off = 0;
  auto alloc = [&](size_t bytes) -> char* {
    char* p = ws + off;
    off += (bytes + 255) & ~(size_t)255;
    return p;
  };
  bf16_t* wqBf    = (bf16_t*)alloc((size_t)768 * CDIM * 2);
  float*  lk      = (float*) alloc((size_t)BATCH * CDIM * 4);
  float*  lq      = (float*) alloc((size_t)BATCH * CDIM * 4);
  bf16_t* xT      = (bf16_t*)alloc((size_t)BATCH * NSP * CDIM * 2);
  float*  qRaw    = (float*) alloc((size_t)BATCH * NSP * CDIM * 4);
  float*  kRaw    = (float*) alloc((size_t)BATCH * CDIM * NSP * 4);
  bf16_t* vBf     = (bf16_t*)alloc((size_t)BATCH * CDIM * NSP * 2);
  bf16_t* kSm     = (bf16_t*)alloc((size_t)BATCH * CDIM * NSP * 2);
  float*  ctx     = (float*) alloc((size_t)BATCH * N_HEADS * D_HEAD * D_HEAD * 4);
  bf16_t* W2      = (bf16_t*)alloc((size_t)BATCH * CDIM * CDIM * 2);
  float*  partial = (float*) alloc((size_t)4096 * 2 * 4);
  float*  stats   = (float*) alloc((size_t)BATCH * 2 * 4);
  // Lifetime-safe aliases (xT dead after K2; qRaw dead after K3):
  bf16_t* qSm = xT;
  float*  y   = qRaw;

  convert_wqkv_kernel<<<768, 256, 0, stream>>>(w_qkv, wqBf, 768 * CDIM);
  lin_kernel<<<32, 256, 0, stream>>>(cond, w_lk, b_lk, w_lq, b_lq, lk, lq);
  transpose_kernel<<<dim3(128, 8, BATCH), 256, 0, stream>>>(x, xT);
  qkv_gemm_kernel<<<dim3(64, 12, BATCH), 128, 0, stream>>>(wqBf, xT, lq, lk,
                                                           qRaw, kRaw, vBf);
  softmax_q_kernel<<<2048, 256, 0, stream>>>(qRaw, qSm);
  softmax_k_kernel<<<4096, 256, 0, stream>>>(kRaw, kSm);
  context_kernel<<<128, 128, 0, stream>>>(kSm, vBf, ctx);
  w2_kernel<<<4096, 256, 0, stream>>>(w_out, ctx, W2);
  out_gemm_kernel<<<dim3(64, 4, BATCH), 128, 0, stream>>>(W2, qSm, b_out, y, partial);
  stats_kernel<<<16, 256, 0, stream>>>(partial, stats);
  norm_kernel<<<65536, 256, 0, stream>>>(y, stats, gamma, beta, out);
}
